// MultiHeadAttention_37452114821425
// MI455X (gfx1250) — matrix operand order repaired and checked by reading, not compiled, not run on hardware
//
#include <hip/hip_runtime.h>

// ---------------------------------------------------------------------------
// MI455X (gfx1250) multi-head attention, all matmuls on v_wmma_f32_16x16x32_bf16
// Softmax runs in the exp2 domain: log2(e)/sqrt(D) is folded into the Q
// projection, so every exponential is a single v_exp_f32.
// ---------------------------------------------------------------------------

typedef __attribute__((ext_vector_type(8)))  float  v8f;
typedef __attribute__((ext_vector_type(8)))  __bf16 v8bf;
typedef __attribute__((ext_vector_type(16))) __bf16 v16bf;

#define DEV __device__ __forceinline__

constexpr int B_  = 2;
constexpr int NQ_ = 2048;
constexpr int NK_ = 2048;
constexpr int E_  = 1024;
constexpr int NH_ = 16;
constexpr int D_  = 64;
constexpr int NTOK_ = 2048;          // NQ_ == NK_; compile-time for div/mod
constexpr float LOG2E   = 1.4426950408889634f;
constexpr float NEGBIG2 = -1.0e8f * LOG2E;   // mask fill, base-2 domain

// --- WMMA fragment helpers --------------------------------------------------
// A (16x32, bf16): lane m = lane&15; half=lane>>4 selects K sub-block.
//   v16bf elems 0..7  = row[kk + half*8 + 0..7]
//   v16bf elems 8..15 = row[kk + 16 + half*8 + 0..7]
DEV v16bf a_frag_bf16(const __bf16* row, int kk, int half) {
    v8bf lo = *(const v8bf*)(row + kk + half * 8);
    v8bf hi = *(const v8bf*)(row + kk + 16 + half * 8);
    v16bf a;
#pragma unroll
    for (int i = 0; i < 8; ++i) { a[i] = lo[i]; a[i + 8] = hi[i]; }
    return a;
}
// B (32x16, bf16): lane n = lane&15; same K order as A:
//   elems 0..7 = K=kk+half*8+0..7, elems 8..15 = K=kk+16+half*8+0..7 (column-major src).
DEV v16bf b_frag(const __bf16* col_base, int kk, int half) {
    v8bf lo = *(const v8bf*)(col_base + kk + half * 8);
    v8bf hi = *(const v8bf*)(col_base + kk + 16 + half * 8);
    v16bf b;
#pragma unroll
    for (int i = 0; i < 8; ++i) { b[i] = lo[i]; b[i + 8] = hi[i]; }
    return b;
}

DEV v8f wmma_bf16(v16bf a, v16bf b, v8f c) {
    return __builtin_amdgcn_wmma_f32_16x16x32_bf16(false, a, false, b,
                                                   (short)0, c, false, false);
}

// --- f32 -> bf16 activation convert -----------------------------------------
__global__ __launch_bounds__(256) void convert_x_kernel(
    const float* __restrict__ x, __bf16* __restrict__ xb) {
    int idx = blockIdx.x * 256 + threadIdx.x;
    xb[idx] = (__bf16)x[idx];
}

// --- weight repack kernels ---------------------------------------------------
// w (NH,E,D) f32 -> Wt (NH*D, E) bf16 : Wt[h*64+d][e] = w[h][e][d]
__global__ __launch_bounds__(256) void convert_wqkv_kernel(
    const float* __restrict__ w, __bf16* __restrict__ wt) {
    int idx = blockIdx.x * 256 + threadIdx.x;         // NH*D*E = 1<<20 threads
    int e  = idx % E_;
    int hd = idx / E_;
    int h  = hd / D_;
    int d  = hd % D_;
    wt[idx] = (__bf16)w[((size_t)h * E_ + e) * D_ + d];
}
// w_o (NH,D,E) f32 -> Wo_t (E, NH*D) bf16 : Wo_t[e][h*64+d] = w_o[h][d][e]
__global__ __launch_bounds__(256) void convert_wo_kernel(
    const float* __restrict__ w, __bf16* __restrict__ wt) {
    int idx = blockIdx.x * 256 + threadIdx.x;         // E*NH*D = 1<<20 threads
    int hd = idx % (NH_ * D_);
    int e  = idx / (NH_ * D_);
    wt[idx] = (__bf16)w[(size_t)hd * E_ + e];
}

// --- QKV projection GEMM -----------------------------------------------------
// Xb (B*NTOK, E) bf16  x  Wt (NH*D, E bf16, col-major-for-B)  ->  bf16 out
// MODE 0: Out[((b*NH+h)*NTOK + t)*64 + d]          (Q, K)
// MODE 1: Out[((b*NH+h)*64 + d)*NTOK + t]          (V transposed)
template <int MODE>
__global__ __launch_bounds__(128) void proj_gemm_kernel(
    const __bf16* __restrict__ Xb, const __bf16* __restrict__ Wt,
    __bf16* __restrict__ Out, float scale) {
    const int lane = threadIdx.x & 31;
    const int wv   = threadIdx.x >> 5;
    const int half = lane >> 4;
    const int l15  = lane & 15;
    const int nTN  = (NH_ * D_) / 16;                 // 64
    int tile = blockIdx.x * 4 + wv;
    int tM = tile / nTN;
    int tN = tile % nTN;

    const __bf16* arow = Xb + (size_t)(tM * 16 + l15) * E_;
    const __bf16* bcol = Wt + (size_t)(tN * 16 + l15) * E_;

    v8f acc = {};
#pragma unroll 2
    for (int kk = 0; kk < E_; kk += 32) {
        v16bf a = a_frag_bf16(arow, kk, half);
        v16bf b = b_frag(bcol, kk, half);
        acc = wmma_bf16(a, b, acc);
    }
    int col = tN * 16 + l15;
    int h = col >> 6, d = col & 63;
#pragma unroll
    for (int v = 0; v < 8; ++v) {
        int t  = tM * 16 + v + 8 * half;
        int bb = t / NTOK_, i = t % NTOK_;            // compile-time -> shifts
        float val = acc[v] * scale;
        if (MODE == 0)
            Out[(size_t)((bb * NH_ + h) * NTOK_ + i) * D_ + d] = (__bf16)val;
        else
            Out[(size_t)((bb * NH_ + h) * D_ + d) * NTOK_ + i] = (__bf16)val;
    }
}

// --- flash attention core ----------------------------------------------------
// Q  [b,h,i,d] bf16 (pre-scaled by log2(e)/sqrt(D))
// K  [b,h,j,d] bf16            (B-frag for S: contiguous along d)
// Vt [b,h,d,j] bf16            (B-frag for PV: contiguous along j)
// O  [b,h,i,d] bf16
// All score/stat math is base-2: p = exp2(s2 - m2); softmax is invariant.
__global__ __launch_bounds__(128) void flash_attn_kernel(
    const __bf16* __restrict__ Q, const __bf16* __restrict__ K,
    const __bf16* __restrict__ Vt, const float* __restrict__ mask,
    __bf16* __restrict__ O) {
    __shared__ __bf16 plds[4][16 * 32];   // per-wave P staging tile (no x-wave use)

    const int lane = threadIdx.x & 31;
    const int wv   = threadIdx.x >> 5;
    const int half = lane >> 4;
    const int l15  = lane & 15;
    const int qb   = blockIdx.x % (NQ_ / 64);
    const int bh   = blockIdx.x / (NQ_ / 64);
    const int b0   = bh / NH_;
    const int i0   = qb * 64 + wv * 16;

    // Q fragments (A layout), loaded once: D=64 -> two K-steps
    const __bf16* qrow = Q + (size_t)(bh * NQ_ + i0 + l15) * D_;
    v16bf qa0 = a_frag_bf16(qrow, 0, half);
    v16bf qa1 = a_frag_bf16(qrow, 32, half);

    v8f o0 = {}, o1 = {}, o2 = {}, o3 = {};
    float mrow[8], lrow[8];
#pragma unroll
    for (int v = 0; v < 8; ++v) { mrow[v] = -3.0e38f; lrow[v] = 0.0f; }

    const __bf16* vbase   = Vt + (size_t)(bh * D_ + l15) * NK_;
    const float*  maskrow = mask + (size_t)(b0 * NQ_ + i0) * NK_;

    for (int j0 = 0; j0 < NK_; j0 += 32) {
        const __bf16* kc0 = K + (size_t)(bh * NK_ + j0 + l15) * D_;
        const __bf16* kc1 = kc0 + 16 * D_;
        if (j0 + 32 < NK_) {                          // hint next key block
            __builtin_prefetch(kc0 + 32 * D_, 0, 0);
            __builtin_prefetch(vbase + j0 + 32, 0, 0);
        }
        // ---- S = Q K^T (two 16x16 tiles, K-loop over D) ----
        v8f s0 = {}, s1 = {};
        s0 = wmma_bf16(qa0, b_frag(kc0, 0, half), s0);
        s0 = wmma_bf16(qa1, b_frag(kc0, 32, half), s0);
        s1 = wmma_bf16(qa0, b_frag(kc1, 0, half), s1);
        s1 = wmma_bf16(qa1, b_frag(kc1, 32, half), s1);

        // ---- mask + online softmax (C layout: row m = v+8*half, col = l15) --
        // s' = NEG + m*(s - NEG)  : one sub + one fma per element
#pragma unroll
        for (int v = 0; v < 8; ++v) {
            const float* mp = maskrow + (size_t)(v + 8 * half) * NK_ + j0;
            float m0 = mp[l15];
            float m1 = mp[16 + l15];
            s0[v] = fmaf(m0, s0[v] - NEGBIG2, NEGBIG2);
            s1[v] = fmaf(m1, s1[v] - NEGBIG2, NEGBIG2);
        }
#pragma unroll
        for (int v = 0; v < 8; ++v) {
            float rmax = fmaxf(s0[v], s1[v]);
#pragma unroll
            for (int off = 1; off < 16; off <<= 1)
                rmax = fmaxf(rmax, __shfl_xor(rmax, off, 32));
            float mn = fmaxf(mrow[v], rmax);
            float sc = exp2f(mrow[v] - mn);           // single v_exp_f32
            mrow[v] = mn;
            float p0 = exp2f(s0[v] - mn);
            float p1 = exp2f(s1[v] - mn);
            float rs = p0 + p1;
#pragma unroll
            for (int off = 1; off < 16; off <<= 1)
                rs += __shfl_xor(rs, off, 32);
            lrow[v] = lrow[v] * sc + rs;
            o0[v] *= sc; o1[v] *= sc; o2[v] *= sc; o3[v] *= sc;
            // P (C layout) -> LDS row-major [m][j] for A-layout reload.
            // Per-wave tile; wave-level LDS ops are in-order, compiler
            // inserts the s_wait_dscnt for the dependent reload below.
            plds[wv][(v + 8 * half) * 32 + l15]      = (__bf16)p0;
            plds[wv][(v + 8 * half) * 32 + 16 + l15] = (__bf16)p1;
        }
        v16bf pa = a_frag_bf16(&plds[wv][l15 * 32], 0, half);
        // ---- O += P V (4 d-groups of 16) ----
        o0 = wmma_bf16(pa, b_frag(vbase + (size_t)0 * 16 * NK_ + j0, 0, half), o0);
        o1 = wmma_bf16(pa, b_frag(vbase + (size_t)1 * 16 * NK_ + j0, 0, half), o1);
        o2 = wmma_bf16(pa, b_frag(vbase + (size_t)2 * 16 * NK_ + j0, 0, half), o2);
        o3 = wmma_bf16(pa, b_frag(vbase + (size_t)3 * 16 * NK_ + j0, 0, half), o3);
    }

    // ---- normalize and store O ----
#pragma unroll
    for (int v = 0; v < 8; ++v) {
        float inv = 1.0f / lrow[v];
        int i = i0 + v + 8 * half;
        __bf16* orow = O + (size_t)(bh * NQ_ + i) * D_;
        orow[0 * 16 + l15] = (__bf16)(o0[v] * inv);
        orow[1 * 16 + l15] = (__bf16)(o1[v] * inv);
        orow[2 * 16 + l15] = (__bf16)(o2[v] * inv);
        orow[3 * 16 + l15] = (__bf16)(o3[v] * inv);
    }
}

// --- output projection: out[b,i,e] = sum_{h,d} O[b,h,i,d] * Wo_t[e][h*64+d] --
__global__ __launch_bounds__(128) void out_proj_kernel(
    const __bf16* __restrict__ Ob, const __bf16* __restrict__ Wot,
    float* __restrict__ out) {
    const int lane = threadIdx.x & 31;
    const int wv   = threadIdx.x >> 5;
    const int half = lane >> 4;
    const int l15  = lane & 15;
    const int nTN  = E_ / 16;                         // 64
    int tile = blockIdx.x * 4 + wv;
    int tM = tile / nTN;
    int tN = tile % nTN;

    int t  = tM * 16 + l15;
    int bb = t / NQ_, i = t % NQ_;                    // compile-time constants
    const __bf16* obase = Ob + ((size_t)(bb * NH_) * NQ_ + i) * D_;
    const __bf16* bcol  = Wot + (size_t)(tN * 16 + l15) * (NH_ * D_);

    v8f acc = {};
#pragma unroll 2
    for (int kk = 0; kk < NH_ * D_; kk += 32) {
        int K0 = kk + half * 8;
        int K1 = kk + 16 + half * 8;
        v8bf lo = *(const v8bf*)(obase + (size_t)(K0 >> 6) * NQ_ * D_ + (K0 & 63));
        v8bf hi = *(const v8bf*)(obase + (size_t)(K1 >> 6) * NQ_ * D_ + (K1 & 63));
        v16bf a;
#pragma unroll
        for (int q = 0; q < 8; ++q) { a[q] = lo[q]; a[q + 8] = hi[q]; }
        v16bf b = b_frag(bcol, kk, half);
        acc = wmma_bf16(a, b, acc);
    }
#pragma unroll
    for (int v = 0; v < 8; ++v) {
        int trow = tM * 16 + v + 8 * half;
        out[(size_t)trow * E_ + tN * 16 + l15] = acc[v];
    }
}

// ---------------------------------------------------------------------------
extern "C" void kernel_launch(void* const* d_in, const int* in_sizes, int n_in,
                              void* d_out, int out_size, void* d_ws, size_t ws_size,
                              hipStream_t stream) {
    const float* key   = (const float*)d_in[0];
    const float* query = (const float*)d_in[1];
    const float* mask  = (const float*)d_in[2];
    const float* w_q   = (const float*)d_in[3];
    const float* w_k   = (const float*)d_in[4];
    const float* w_v   = (const float*)d_in[5];
    const float* w_o   = (const float*)d_in[6];
    float* out = (float*)d_out;

    char* ws = (char*)d_ws;
    size_t off = 0;
    auto alloc = [&](size_t bytes) {
        void* p = ws + off;
        off += (bytes + 255) & ~(size_t)255;
        return p;
    };
    const size_t qkvElems = (size_t)B_ * NH_ * NQ_ * D_;   // 4M
    const size_t wElems   = (size_t)NH_ * D_ * E_;         // 1M
    const size_t xElems   = (size_t)B_ * NTOK_ * E_;       // 4M
    __bf16* Qb   = (__bf16*)alloc(qkvElems * 2);
    __bf16* Kb   = (__bf16*)alloc(qkvElems * 2);
    __bf16* Vt   = (__bf16*)alloc(qkvElems * 2);
    __bf16* Obuf = (__bf16*)alloc(qkvElems * 2);
    __bf16* Wqt  = (__bf16*)alloc(wElems * 2);
    __bf16* Wkt  = (__bf16*)alloc(wElems * 2);
    __bf16* Wvt  = (__bf16*)alloc(wElems * 2);
    __bf16* Wot  = (__bf16*)alloc(wElems * 2);
    __bf16* Xq   = (__bf16*)alloc(xElems * 2);
    __bf16* Xk   = (__bf16*)alloc(xElems * 2);
    (void)in_sizes; (void)n_in; (void)out_size; (void)ws_size;

    // 1) convert activations + repack weights to bf16 (WMMA-friendly layouts)
    const int xBlocks   = (int)(xElems / 256);             // 16384
    const int cvtBlocks = (int)(wElems / 256);             // 4096
    convert_x_kernel   <<<xBlocks,   256, 0, stream>>>(query, Xq);
    convert_x_kernel   <<<xBlocks,   256, 0, stream>>>(key,   Xk);
    convert_wqkv_kernel<<<cvtBlocks, 256, 0, stream>>>(w_q, Wqt);
    convert_wqkv_kernel<<<cvtBlocks, 256, 0, stream>>>(w_k, Wkt);
    convert_wqkv_kernel<<<cvtBlocks, 256, 0, stream>>>(w_v, Wvt);
    convert_wo_kernel  <<<cvtBlocks, 256, 0, stream>>>(w_o, Wot);

    // 2) Q/K/V projections (WMMA GEMMs).
    //    Fold log2(e)/sqrt(D) into Q so softmax runs in the exp2 domain.
    const int projBlocks = ((B_ * NTOK_ / 16) * (NH_ * D_ / 16)) / 4;  // 4096
    proj_gemm_kernel<0><<<projBlocks, 128, 0, stream>>>(Xq, Wqt, Qb, 0.125f * LOG2E);
    proj_gemm_kernel<0><<<projBlocks, 128, 0, stream>>>(Xk, Wkt, Kb, 1.0f);
    proj_gemm_kernel<1><<<projBlocks, 128, 0, stream>>>(Xk, Wvt, Vt, 1.0f);

    // 3) flash attention core
    const int faBlocks = B_ * NH_ * (NQ_ / 64);            // 1024
    flash_attn_kernel<<<faBlocks, 128, 0, stream>>>(Qb, Kb, Vt, mask, Obuf);

    // 4) output projection
    const int outBlocks = ((B_ * NQ_ / 16) * (E_ / 16)) / 4;  // 4096
    out_proj_kernel<<<outBlocks, 128, 0, stream>>>(Obuf, Wot, out);
}
